// RandomPermutationSparseTokenizer_72103910965632
// MI455X (gfx1250) — compile-verified, matrix operand order checked by reading, not run on hardware
//
#include <hip/hip_runtime.h>

// ---------------------------------------------------------------------------
// RandomPermutationSparseTokenizer for MI455X (gfx1250, wave32, WMMA)
//   K1: embedding gather + mean pool -> X f16 [131072][128]   (memory bound)
//   K2: X @ W^T + b -> SiLU -> LayerNorm using v_wmma_f32_16x16x32_f16
// ---------------------------------------------------------------------------

typedef __attribute__((ext_vector_type(16))) _Float16 v16h;
typedef __attribute__((ext_vector_type(8)))  _Float16 v8h;
typedef __attribute__((ext_vector_type(8)))  float    v8f;

#define B_     8192
#define F_     64
#define V_     1000
#define E_     64
#define T_     2
#define NTOK_  16
#define G_     4
#define D_     256
#define K_     128                 // = E_ * T_
#define MROWS  (B_ * NTOK_)        // 131072 token rows

#define MT     64                  // rows (tokens) per workgroup in GEMM
#define WKP    136                 // padded K stride for W in LDS (halves; 272B -> 16B aligned rows)
#define HSTR   257                 // padded row stride for h in LDS (floats; bank-conflict-free)

// ---------------- Kernel 1: gather + missing-add + mean pool ----------------
// 2 tokens per 256-thread block; 128 threads per token (t = c>>6, e = c&63).
// Lane-consecutive `e` -> fully coalesced 256B reads of each embedding row.
__global__ __launch_bounds__(256) void tok_gather_pool(
    const int* __restrict__ ids, const int* __restrict__ mmask,
    const int* __restrict__ gidx, const float* __restrict__ tbl,
    const float* __restrict__ miss, _Float16* __restrict__ x)
{
  const int tid   = threadIdx.x;
  const int token = blockIdx.x * 2 + (tid >> 7);
  const int c     = tid & 127;          // output channel 0..127
  const int t     = c >> 6;             // table id
  const int e     = c & 63;             // embedding element
  const int b     = token >> 4;         // batch row
  const int n     = token & 15;         // token-group id

  float sum = 0.f;
#pragma unroll
  for (int g = 0; g < G_; ++g) {
    const int   f  = gidx[n * G_ + g];
    const int   id = ids[b * F_ + f];
    const float m  = (float)mmask[b * F_ + f];
    const size_t toff = (((size_t)t * F_ + f) * V_ + (size_t)id) * E_ + e;
    sum += tbl[toff] + m * miss[((t * F_) + f) * E_ + e];
  }
  x[(size_t)token * K_ + c] = (_Float16)(sum * 0.25f);
}

// ------------- Kernel 2: GEMM (WMMA f16->f32) + bias + SiLU + LN ------------
__global__ __launch_bounds__(128) void gemm_silu_ln(
    const _Float16* __restrict__ x, const float* __restrict__ Wm,
    const float* __restrict__ bias, const float* __restrict__ gamma,
    const float* __restrict__ beta, float* __restrict__ out)
{
  extern __shared__ char smem[];
  _Float16* w_lds  = (_Float16*)smem;                                      // [D_][WKP]
  float*    h_lds  = (float*)(smem + (size_t)D_ * WKP * sizeof(_Float16)); // [MT][HSTR]
  float*    st_lds = h_lds + MT * HSTR;                                    // mu[MT], rsig[MT]

  const int tid   = threadIdx.x;
  const int wave  = tid >> 5;
  const int lane  = tid & 31;
  const int mbase = blockIdx.x * MT;

  // Stage W (f32 -> f16) into LDS, padded rows (conflict-free ds_load_b128).
  for (int i = tid; i < D_ * K_; i += 128) {
    const int nrow = i >> 7, k = i & 127;
    w_lds[nrow * WKP + k] = (_Float16)Wm[i];
  }

  // Preload this lane's 16 bias values (column lane&15 of each N-tile) so the
  // loads overlap with the WMMA pipeline instead of stalling the epilogue.
  const int col0 = lane & 15;
  float bb[16];
#pragma unroll
  for (int nt = 0; nt < 16; ++nt) bb[nt] = bias[16 * nt + col0];

  __syncthreads();

  // A fragments for this wave's 16-row M-tile, all of K=128 (4 k-steps).
  // ISA 16-bit A 16x32 layout: lanes 0-15 hold K {0..7,16..23}, lanes 16-31 hold K {8..15,24..31}.
  union Frag { v16h v; struct { v8h lo, hi; } p; };
  const int arow = mbase + wave * 16 + (lane & 15);
  const int koff = (lane < 16) ? 0 : 8;
  const _Float16* xr = x + (size_t)arow * K_;
  Frag a[4];
#pragma unroll
  for (int kt = 0; kt < 4; ++kt) {
    a[kt].p.lo = *(const v8h*)(xr + 32 * kt + koff);       // 16B aligned global_load_b128
    a[kt].p.hi = *(const v8h*)(xr + 32 * kt + koff + 16);
  }

  // C layout: VGPR r <-> local row (wave*16 + r + 8*(lane>=16)), col = lane&15.
  const int lrowbase = wave * 16 + 8 * (lane >> 4);

#pragma unroll
  for (int nt = 0; nt < 16; ++nt) {
    v8f acc = {};
    const _Float16* wp = &w_lds[(16 * nt + col0) * WKP];
#pragma unroll
    for (int kt = 0; kt < 4; ++kt) {
      Frag bf;
      bf.p.lo = *(const v8h*)(wp + 32 * kt + koff);        // ds_load_b128
      bf.p.hi = *(const v8h*)(wp + 32 * kt + koff + 16);
      acc = __builtin_amdgcn_wmma_f32_16x16x32_f16(
          /*neg_a=*/false, a[kt].v, /*neg_b=*/false, bf.v,
          /*c_mod=*/(short)0, acc, /*reuse_a=*/false, /*reuse_b=*/false);
    }
#pragma unroll
    for (int r = 0; r < 8; ++r) {
      float v = acc[r] + bb[nt];
      // SiLU with HW reciprocal (v_rcp_f32) instead of IEEE div sequence.
      v = v * __builtin_amdgcn_rcpf(1.f + __expf(-v));
      h_lds[(lrowbase + r) * HSTR + 16 * nt + col0] = v;
    }
  }
  __syncthreads();

  // Per-row LayerNorm stats (one thread per row; padded stride -> no conflicts).
  if (tid < MT) {
    const float* hr = h_lds + tid * HSTR;
    float s = 0.f, s2 = 0.f;
#pragma unroll 8
    for (int i = 0; i < D_; ++i) { const float v = hr[i]; s += v; s2 += v * v; }
    const float mu  = s * (1.f / D_);
    const float var = s2 * (1.f / D_) - mu * mu;
    st_lds[tid]      = mu;
    st_lds[MT + tid] = __builtin_amdgcn_rsqf(var + 1e-5f);  // v_rsq_f32
  }
  __syncthreads();

  // Normalize + coalesced store.
  for (int i = tid; i < MT * D_; i += 128) {
    const int r = i >> 8, cc = i & 255;
    const float v = h_lds[r * HSTR + cc];
    out[(size_t)(mbase + r) * D_ + cc] =
        (v - st_lds[r]) * st_lds[MT + r] * gamma[cc] + beta[cc];
  }
}

// ---------------------------------------------------------------------------
extern "C" void kernel_launch(void* const* d_in, const int* in_sizes, int n_in,
                              void* d_out, int out_size, void* d_ws, size_t ws_size,
                              hipStream_t stream)
{
  (void)in_sizes; (void)n_in; (void)out_size; (void)ws_size;
  const int*   ids   = (const int*)d_in[0];   // int_feats   (B,F)
  const int*   mmask = (const int*)d_in[1];   // missing_mask(B,F)
  const int*   gidx  = (const int*)d_in[2];   // group_idx   (NTOK,G)
  const float* tbl   = (const float*)d_in[3]; // emb_tables  (T,F,V,E)
  const float* miss  = (const float*)d_in[4]; // missing_emb (T,F,E)
  const float* Wm    = (const float*)d_in[5]; // W           (D, E*T)
  const float* bias  = (const float*)d_in[6]; // b           (D,)
  const float* gam   = (const float*)d_in[7]; // ln_gamma    (D,)
  const float* bet   = (const float*)d_in[8]; // ln_beta     (D,)
  float*    out = (float*)d_out;              // (B, NTOK, D) f32
  _Float16* xws = (_Float16*)d_ws;            // 131072*128 f16 = 33.5 MB

  tok_gather_pool<<<MROWS / 2, 256, 0, stream>>>(ids, mmask, gidx, tbl, miss, xws);

  const size_t smem = (size_t)D_ * WKP * sizeof(_Float16)   // W f16:  69632 B
                    + (size_t)MT * HSTR * sizeof(float)     // h tile: 65792 B
                    + (size_t)2 * MT * sizeof(float);       // stats:    512 B
  gemm_silu_ln<<<MROWS / MT, 128, smem, stream>>>(xws, Wm, bias, gam, bet, out);
}
